// GATSingleHeadLayerEdgeReprFeat_11914239279934
// MI455X (gfx1250) — matrix operand order repaired and checked by reading, not compiled, not run on hardware
//
#include <hip/hip_runtime.h>
#include <hip/hip_bf16.h>

typedef unsigned short u16;
typedef __attribute__((ext_vector_type(16))) __bf16 v16bf;
typedef __attribute__((ext_vector_type(8)))  __bf16 v8bf;
typedef __attribute__((ext_vector_type(8)))  float  v8f;

// ---------- helpers ----------

__device__ __forceinline__ u16 f2bf_u(float f) {
    __hip_bfloat16 b = __float2bfloat16(f);     // HW cvt on gfx1250
    return __builtin_bit_cast(u16, b);
}
__device__ __forceinline__ float bf2f(u16 h) {
    unsigned u = ((unsigned)h) << 16;
    return __builtin_bit_cast(float, u);
}

// A operand (16xK tile, row-major bf16 in LDS, leading dim ldk).
// 16-bit A layout: lanes 0-15 row M=l hold K {kb+0..7, kb+16..23};
// lanes 16-31 row M=l-16 hold K {kb+8..15, kb+24..31}.
__device__ __forceinline__ v16bf load_a(const u16* tile, int ldk, int kbase, int lane) {
    int row = lane & 15;
    int kb  = kbase + ((lane & 16) ? 8 : 0);
    const u16* p = tile + row * ldk + kb;
    v8bf lo = *reinterpret_cast<const v8bf*>(p);        // K = kb .. kb+7
    v8bf hi = *reinterpret_cast<const v8bf*>(p + 16);   // K = kb+16 .. kb+23
    return __builtin_shufflevector(lo, hi, 0,1,2,3,4,5,6,7,8,9,10,11,12,13,14,15);
}

// B operand (KxN tile). Weights stored TRANSPOSED in LDS: wt[n*ldk + k].
// 16-bit B layout: lanes 0-15 col N=l hold K {kb+0..7, kb+16..23}; lanes 16-31 hold K {kb+8..15, kb+24..31}.
__device__ __forceinline__ v16bf load_b(const u16* wt, int ldk, int ntile, int kbase, int lane) {
    int n  = ntile * 16 + (lane & 15);
    int kb = kbase + ((lane & 16) ? 8 : 0);
    const u16* p = wt + n * ldk + kb;
    v8bf lo = *reinterpret_cast<const v8bf*>(p);
    v8bf hi = *reinterpret_cast<const v8bf*>(p + 16);
    return __builtin_shufflevector(lo, hi, 0,1,2,3,4,5,6,7,8,9,10,11,12,13,14,15);
}

#define WMMA_BF16(c, a, b) \
    __builtin_amdgcn_wmma_f32_16x16x32_bf16(false, (a), false, (b), (short)0, (c), false, false)

// ---------- kernel 0: init accumulators ----------

__global__ void init_kernel(float* __restrict__ hout, int* __restrict__ m_i,
                            float* __restrict__ s, int N) {
    int i = blockIdx.x * blockDim.x + threadIdx.x;
    if (i < N * 64) hout[i] = 0.f;
    if (i < N) { m_i[i] = 0; s[i] = 0.f; }
}

// ---------- kernel 1/2: X[rows,64] -> relu(X W1) W2, W1,W2: [64,64] row-major ----------
// 4 waves/block, one 16-row tile per wave. Outputs f32 and/or bf16.

__global__ __launch_bounds__(128) void mlp64_kernel(
    const float* __restrict__ X, const float* __restrict__ W1, const float* __restrict__ W2,
    float* __restrict__ Yf, u16* __restrict__ Ybf, int rows)
{
    __shared__ __align__(16) u16 wt1[64 * 64];
    __shared__ __align__(16) u16 wt2[64 * 64];
    __shared__ __align__(16) u16 stage[4][16 * 64];
    __shared__ __align__(16) u16 hid[4][16 * 64];

    const int tid = threadIdx.x, lane = tid & 31, w = tid >> 5;

    // weights -> LDS, transposed, bf16
    for (int i = tid; i < 64 * 64; i += 128) {
        int k = i >> 6, n = i & 63;
        wt1[n * 64 + k] = f2bf_u(W1[i]);
        wt2[n * 64 + k] = f2bf_u(W2[i]);
    }
    __syncthreads();

    const long r0 = (long)blockIdx.x * 64 + w * 16;

    // stage 16x64 input tile: coalesced b128 loads, f32 -> bf16, b64 LDS stores
    {
        const float4* X4 = reinterpret_cast<const float4*>(X);
        const long base4 = r0 * 16;                  // 16 float4 per row
        const long lim4  = (long)rows * 16;
        for (int i = lane; i < 256; i += 32) {
            long idx4 = base4 + i;
            float4 f = (idx4 < lim4) ? X4[idx4] : make_float4(0.f, 0.f, 0.f, 0.f);
            unsigned lo = (unsigned)f2bf_u(f.x) | ((unsigned)f2bf_u(f.y) << 16);
            unsigned hi = (unsigned)f2bf_u(f.z) | ((unsigned)f2bf_u(f.w) << 16);
            uint2 pk; pk.x = lo; pk.y = hi;
            *reinterpret_cast<uint2*>(&stage[w][i * 4]) = pk;
        }
    }
    __syncthreads();

    // GEMM1 (16x64 @ 64x64) + relu -> hid
    for (int t = 0; t < 4; ++t) {
        v8f c = {};
        for (int ch = 0; ch < 2; ++ch) {
            v16bf a = load_a(stage[w], 64, ch * 32, lane);
            v16bf b = load_b(wt1, 64, t, ch * 32, lane);
            c = WMMA_BF16(c, a, b);
        }
        int n = t * 16 + (lane & 15);
        int rbase = (lane & 16) ? 8 : 0;
        #pragma unroll
        for (int r = 0; r < 8; ++r)
            hid[w][(rbase + r) * 64 + n] = f2bf_u(fmaxf(c[r], 0.f));
    }
    __syncthreads();

    // GEMM2 (16x64 @ 64x64) -> outputs (no activation on z)
    for (int t = 0; t < 4; ++t) {
        v8f c = {};
        for (int ch = 0; ch < 2; ++ch) {
            v16bf a = load_a(hid[w], 64, ch * 32, lane);
            v16bf b = load_b(wt2, 64, t, ch * 32, lane);
            c = WMMA_BF16(c, a, b);
        }
        int n = t * 16 + (lane & 15);
        int rbase = (lane & 16) ? 8 : 0;
        #pragma unroll
        for (int r = 0; r < 8; ++r) {
            long row = r0 + rbase + r;
            if (row < rows) {
                float v = c[r];
                if (Yf)  Yf[row * 64 + n] = v;
                if (Ybf) Ybf[row * 64 + n] = f2bf_u(v);
            }
        }
    }
}

// ---------- kernel 3: per-edge concat -> proj MLP (bias) + attention score ----------
// z = [z_e | z_h[src] | z_h[dst]] (16x192 bf16 tile per wave)
// e_proj = relu( relu(z Wp1 + bp1) Wp2 + bp2 ) ; a = relu(z . Wa)

__global__ __launch_bounds__(128) void edge_proj_attn_kernel(
    const u16* __restrict__ zebf, const u16* __restrict__ zhbf,
    const int* __restrict__ src, const int* __restrict__ dst,
    const float* __restrict__ Wp1, const float* __restrict__ bp1,
    const float* __restrict__ Wp2, const float* __restrict__ bp2,
    const float* __restrict__ Wa,
    float* __restrict__ eout, float* __restrict__ a_out, int E)
{
    __shared__ __align__(16) u16 wtp1[64 * 192];   // transposed [n][k]
    __shared__ __align__(16) u16 wtp2[64 * 64];
    __shared__ float b1s[64], b2s[64], was[192];
    __shared__ __align__(16) u16 zst[4][16 * 192];
    __shared__ __align__(16) u16 hid[4][16 * 64];

    const int tid = threadIdx.x, lane = tid & 31, w = tid >> 5;

    for (int i = tid; i < 192 * 64; i += 128) {
        int k = i / 64, n = i % 64;               // Wp1 is (192,64) row-major
        wtp1[n * 192 + k] = f2bf_u(Wp1[i]);
    }
    for (int i = tid; i < 64 * 64; i += 128) {
        int k = i >> 6, n = i & 63;
        wtp2[n * 64 + k] = f2bf_u(Wp2[i]);
    }
    if (tid < 64) { b1s[tid] = bp1[tid]; b2s[tid] = bp2[tid]; }
    for (int i = tid; i < 192; i += 128) was[i] = Wa[i];
    __syncthreads();

    const int e0 = blockIdx.x * 64 + w * 16;
    const uint4* ze4 = reinterpret_cast<const uint4*>(zebf);  // 8 bf16 per uint4
    const uint4* zh4 = reinterpret_cast<const uint4*>(zhbf);

    // stage z_e rows (contiguous) into cols 0..63 : b128 copies
    for (int i = lane; i < 128; i += 32) {
        int row = i >> 3, d8 = i & 7;
        int er = e0 + row; if (er >= E) er = E - 1;
        *reinterpret_cast<uint4*>(&zst[w][row * 192 + d8 * 8]) = ze4[(long)er * 8 + d8];
    }
    // gather z_h[src] -> cols 64..127, z_h[dst] -> cols 128..191 : b128 row gathers
    for (int i = lane; i < 128; i += 32) {
        int row = i >> 3, d8 = i & 7;
        int er = e0 + row; if (er >= E) er = E - 1;
        int si = src[er], di = dst[er];
        *reinterpret_cast<uint4*>(&zst[w][row * 192 + 64  + d8 * 8]) = zh4[(long)si * 8 + d8];
        *reinterpret_cast<uint4*>(&zst[w][row * 192 + 128 + d8 * 8]) = zh4[(long)di * 8 + d8];
    }
    __syncthreads();

    // attention score: a = relu(z . Wa). Lane l and l^16 split the 192-dot.
    {
        int row = lane & 15;
        int kb = (lane & 16) ? 96 : 0;
        float acc = 0.f;
        #pragma unroll
        for (int c = 0; c < 12; ++c) {
            uint4 v = *reinterpret_cast<const uint4*>(&zst[w][row * 192 + kb + c * 8]);
            unsigned vv[4] = {v.x, v.y, v.z, v.w};
            #pragma unroll
            for (int j = 0; j < 4; ++j) {
                acc += bf2f((u16)(vv[j] & 0xFFFFu)) * was[kb + c * 8 + 2 * j]
                     + bf2f((u16)(vv[j] >> 16))     * was[kb + c * 8 + 2 * j + 1];
            }
        }
        acc += __shfl_xor(acc, 16, 32);
        if (lane < 16) {
            int er = e0 + row;
            if (er < E) a_out[er] = fmaxf(acc, 0.f);
        }
    }

    // GEMM1: 16x192 @ 192x64, + bias, relu -> hid
    for (int t = 0; t < 4; ++t) {
        v8f c = {};
        for (int ch = 0; ch < 6; ++ch) {
            v16bf a = load_a(zst[w], 192, ch * 32, lane);
            v16bf b = load_b(wtp1, 192, t, ch * 32, lane);
            c = WMMA_BF16(c, a, b);
        }
        int n = t * 16 + (lane & 15);
        float bn = b1s[n];
        int rbase = (lane & 16) ? 8 : 0;
        #pragma unroll
        for (int r = 0; r < 8; ++r)
            hid[w][(rbase + r) * 64 + n] = f2bf_u(fmaxf(c[r] + bn, 0.f));
    }
    __syncthreads();

    // GEMM2: 16x64 @ 64x64, + bias, final relu -> e_proj output
    for (int t = 0; t < 4; ++t) {
        v8f c = {};
        for (int ch = 0; ch < 2; ++ch) {
            v16bf a = load_a(hid[w], 64, ch * 32, lane);
            v16bf b = load_b(wtp2, 64, t, ch * 32, lane);
            c = WMMA_BF16(c, a, b);
        }
        int n = t * 16 + (lane & 15);
        float bn = b2s[n];
        int rbase = (lane & 16) ? 8 : 0;
        #pragma unroll
        for (int r = 0; r < 8; ++r) {
            int er = e0 + rbase + r;
            if (er < E) eout[(long)er * 64 + n] = fmaxf(c[r] + bn, 0.f);
        }
    }
}

// ---------- softmax over incoming edges + weighted scatter ----------

__global__ void seg_max_kernel(const float* __restrict__ a, const int* __restrict__ dst,
                               int* __restrict__ m_i, int E) {
    int e = blockIdx.x * blockDim.x + threadIdx.x;
    if (e < E) {
        // a >= 0, m initialized to 0 -> int compare == float compare; empty segs stay 0
        atomicMax(&m_i[dst[e]], __float_as_int(a[e]));
    }
}

__global__ void exp_sum_kernel(const float* __restrict__ a, const int* __restrict__ dst,
                               const int* __restrict__ m_i, float* __restrict__ ex,
                               float* __restrict__ s, int E) {
    int e = blockIdx.x * blockDim.x + threadIdx.x;
    if (e < E) {
        int de = dst[e];
        float mm = __int_as_float(m_i[de]);
        float x = __expf(a[e] - mm);
        ex[e] = x;
        atomicAdd(&s[de], x);
    }
}

__global__ void scatter_kernel(const float* __restrict__ ex, const float* __restrict__ s,
                               const int* __restrict__ src, const int* __restrict__ dst,
                               const float* __restrict__ zhf, float* __restrict__ hout,
                               long total) {
    long i = (long)blockIdx.x * blockDim.x + threadIdx.x;
    if (i < total) {
        int e = (int)(i >> 6), d = (int)(i & 63);
        int de = dst[e];
        float alpha = ex[e] / fmaxf(s[de], 1e-9f);
        atomicAdd(&hout[(long)de * 64 + d], alpha * zhf[(long)src[e] * 64 + d]);
    }
}

__global__ void relu_kernel(float* __restrict__ p, long n) {
    long i = (long)blockIdx.x * blockDim.x + threadIdx.x;
    if (i < n) p[i] = fmaxf(p[i], 0.f);
}

// ---------- host launcher ----------

static inline size_t align256(size_t x) { return (x + 255) & ~(size_t)255; }

extern "C" void kernel_launch(void* const* d_in, const int* in_sizes, int n_in,
                              void* d_out, int out_size, void* d_ws, size_t ws_size,
                              hipStream_t stream) {
    const float* h   = (const float*)d_in[0];
    const float* e   = (const float*)d_in[1];
    const int*   src = (const int*)  d_in[2];
    const int*   dst = (const int*)  d_in[3];
    const float* Wh1 = (const float*)d_in[4];
    const float* Wh2 = (const float*)d_in[5];
    const float* We1 = (const float*)d_in[6];
    const float* We2 = (const float*)d_in[7];
    const float* Wp1 = (const float*)d_in[8];
    const float* bp1 = (const float*)d_in[9];
    const float* Wp2 = (const float*)d_in[10];
    const float* bp2 = (const float*)d_in[11];
    const float* Wa  = (const float*)d_in[12];

    const int N = in_sizes[0] / 64;
    const int E = in_sizes[2];

    float* hout = (float*)d_out;                 // [N,64]
    float* eout = (float*)d_out + (long)N * 64;  // [E,64]

    // workspace carve
    char* ws = (char*)d_ws;
    size_t off = 0;
    float* zhf = (float*)(ws + off); off = align256(off + (size_t)N * 64 * 4);
    u16*   zhb = (u16*)  (ws + off); off = align256(off + (size_t)N * 64 * 2);
    u16*   zeb = (u16*)  (ws + off); off = align256(off + (size_t)E * 64 * 2);
    float* aEd = (float*)(ws + off); off = align256(off + (size_t)E * 4);
    float* exE = (float*)(ws + off); off = align256(off + (size_t)E * 4);
    int*   mI  = (int*)  (ws + off); off = align256(off + (size_t)N * 4);
    float* sF  = (float*)(ws + off); off = align256(off + (size_t)N * 4);
    (void)ws_size; (void)n_in; (void)out_size;

    // 0. zero accumulators (d_out node region, seg-max, seg-sum)
    init_kernel<<<(N * 64 + 255) / 256, 256, 0, stream>>>(hout, mI, sF, N);

    // 1. z_h = relu(h Wh1) Wh2  -> f32 + bf16
    mlp64_kernel<<<(N + 63) / 64, 128, 0, stream>>>(h, Wh1, Wh2, zhf, zhb, N);

    // 2. z_e = relu(e We1) We2  -> bf16 only
    mlp64_kernel<<<(E + 63) / 64, 128, 0, stream>>>(e, We1, We2, nullptr, zeb, E);

    // 3. per-edge concat -> proj MLP (writes e_proj output) + attention logits
    edge_proj_attn_kernel<<<(E + 63) / 64, 128, 0, stream>>>(
        zeb, zhb, src, dst, Wp1, bp1, Wp2, bp2, Wa, eout, aEd, E);

    // 4-6. segment softmax over dst
    seg_max_kernel<<<(E + 255) / 256, 256, 0, stream>>>(aEd, dst, mI, E);
    exp_sum_kernel<<<(E + 255) / 256, 256, 0, stream>>>(aEd, dst, mI, exE, sF, E);

    // 7. h_out += alpha * z_h[src]
    long total = (long)E * 64;
    scatter_kernel<<<(unsigned)((total + 255) / 256), 256, 0, stream>>>(
        exE, sF, src, dst, zhf, hout, total);

    // 8. final relu on node output
    relu_kernel<<<(N * 64 + 255) / 256, 256, 0, stream>>>(hout, (long)N * 64);
}
